// Encoder_22265110463126
// MI455X (gfx1250) — matrix operand order repaired and checked by reading, not compiled, not run on hardware
//
#include <hip/hip_runtime.h>
#include <math.h>
#include <stdint.h>

typedef unsigned short u16;
typedef unsigned int   u32;
typedef unsigned long long u64;

typedef __attribute__((ext_vector_type(16))) __bf16 v16bf;
typedef __attribute__((ext_vector_type(8)))  float  v8f;

// ---------------- problem dims ----------------
constexpr int BATCH = 16384;
constexpr int OBS   = 512;
constexpr int ACTD  = 64;
constexpr int DIN   = 1090;           // 512+64+512+1+1
constexpr int DINP  = 1120;           // padded to multiple of 32 (WMMA K step)
constexpr int HID   = 4096;
constexpr int DL    = 1024;           // latent dim
constexpr int KC    = 512;            // codebook size
constexpr float DECAY = 0.99f;
constexpr float CCOST = 0.25f;
constexpr float EPSF  = 1e-5f;

// ---------------- GEMM tiling ----------------
constexpr int TILE_M = 128;
constexpr int TILE_N = 128;
constexpr int TILE_K = 32;

// ---------------- TDM (Tensor Data Mover) availability ----------------
#if __has_builtin(__builtin_amdgcn_tensor_load_to_lds)
#define USE_TDM 1
typedef __attribute__((ext_vector_type(4))) unsigned int v4u;
typedef __attribute__((ext_vector_type(8))) int v8i;
typedef __attribute__((ext_vector_type(4))) int v4i;
#else
#define USE_TDM 0
#endif

__device__ __forceinline__ u16 f2bf(float f) {
  u32 u = __builtin_bit_cast(u32, f);
  u += 0x7FFFu + ((u >> 16) & 1u);    // round-to-nearest-even
  return (u16)(u >> 16);
}

#if USE_TDM
// Issue a TDM 2-D tile load: TILE_M rows x TILE_K bf16, row stride Kd elements,
// into contiguous LDS at lds_off. D# packed per CDNA5 ISA §8.3/8.4.
__device__ __forceinline__ void tdm_load_tile_a(const u16* gptr, u32 lds_off, int Kd) {
  const u64 ga  = (u64)(uintptr_t)gptr;
  const u64 td0 = (u64)(u32)Kd;        // tensor_dim0 (elements per row)
  const u64 td1 = (u64)BATCH;          // tensor_dim1 (rows available)
  const u64 st0 = (u64)(u32)Kd;        // tensor_dim0_stride (elements)
  v4u g0;
  g0[0] = 1u;                                               // count=1, user D#
  g0[1] = lds_off;                                          // lds_addr (bytes)
  g0[2] = (u32)(ga & 0xFFFFFFFFull);                        // global_addr[31:0]
  g0[3] = (u32)((ga >> 32) & 0x01FFFFFFull) | (2u << 30);   // [56:32] | type=2
  v8i g1;
  g1[0] = (int)(1u << 16);                                  // data_size=1 -> 2 bytes
  g1[1] = (int)((td0 & 0xFFFFull) << 16);                   // tensor_dim0[15:0] @ bit48
  g1[2] = (int)(((td0 >> 16) & 0xFFFFull) | ((td1 & 0xFFFFull) << 16));
  g1[3] = (int)(((td1 >> 16) & 0xFFFFull) | ((u32)TILE_K << 16));  // tile_dim0
  g1[4] = (int)((u32)TILE_M);                               // tile_dim1 (tile_dim2=0)
  g1[5] = (int)(st0 & 0xFFFFFFFFull);                       // stride0[31:0]
  g1[6] = (int)((st0 >> 32) & 0xFFFFull);                   // stride0[47:32]
  g1[7] = 0;
  v4i z4 = {0, 0, 0, 0};
#if defined(__clang_major__) && __clang_major__ >= 23
  v8i z8 = {0, 0, 0, 0, 0, 0, 0, 0};
  __builtin_amdgcn_tensor_load_to_lds(g0, g1, z4, z4, z8, 0);
#else
  __builtin_amdgcn_tensor_load_to_lds(g0, g1, z4, z4, 0);
#endif
}

__device__ __forceinline__ void tdm_wait0() {
#if __has_builtin(__builtin_amdgcn_s_wait_tensorcnt)
  __builtin_amdgcn_s_wait_tensorcnt(0);
#else
  asm volatile("s_wait_tensorcnt 0x0" ::: "memory");
#endif
}
#endif  // USE_TDM

// ---------------------------------------------------------------------------
// Pack hstack([obs, action, next_obs, reward, termination]) -> bf16 [B, DINP]
// ---------------------------------------------------------------------------
__global__ void pack_x_kernel(const float* __restrict__ obs,
                              const float* __restrict__ act,
                              const float* __restrict__ nobs,
                              const float* __restrict__ rew,
                              const float* __restrict__ term,
                              u16* __restrict__ X) {
  size_t i = (size_t)blockIdx.x * blockDim.x + threadIdx.x;
  size_t n = (size_t)BATCH * DINP;
  if (i >= n) return;
  int b = (int)(i / DINP), c = (int)(i % DINP);
  float v = 0.0f;
  if (c < OBS)                      v = obs[(size_t)b * OBS + c];
  else if (c < OBS + ACTD)          v = act[(size_t)b * ACTD + (c - OBS)];
  else if (c < 2 * OBS + ACTD)      v = nobs[(size_t)b * OBS + (c - OBS - ACTD)];
  else if (c == 2 * OBS + ACTD)     v = rew[b];
  else if (c == 2 * OBS + ACTD + 1) v = term[b];
  X[i] = f2bf(v);
}

// fp32 [Krows,N] weight -> bf16 [Kpad,N] (zero-padded K rows)
__global__ void cvt_w_kernel(const float* __restrict__ W, u16* __restrict__ Wb,
                             int Krows, int Kpad, int N) {
  size_t i = (size_t)blockIdx.x * blockDim.x + threadIdx.x;
  size_t n = (size_t)Kpad * N;
  if (i >= n) return;
  int k = (int)(i / N), c = (int)(i % N);
  Wb[i] = f2bf(k < Krows ? W[(size_t)k * N + c] : 0.0f);
}

// ---------------------------------------------------------------------------
// Tiled bf16 WMMA GEMM:  out = act( A[M,Kd] @ Bw[Kd,N] + bias )
//   block = 256 threads = 8 waves, tile 128x128x32, double-buffered LDS.
//   Wave grid 4(M) x 2(N): each wave owns 32x64 = 2 A-frags x 4 B-frags
//   -> 8 x v_wmma_f32_16x16x32_bf16 per K step.
//   A tile staged by the Tensor Data Mover (tensor_load_to_lds + TENSORcnt)
//   when available; B tile staged transposed by the threads.
// ---------------------------------------------------------------------------
__global__ __launch_bounds__(256) void gemm_bf16_wmma(
    const u16* __restrict__ A, const u16* __restrict__ Bw,
    const float* __restrict__ bias,
    u16* __restrict__ outb, float* __restrict__ outf,
    int M, int Kd, int N, int relu) {
  __shared__ u16 lA[2][TILE_M][TILE_K]     __attribute__((aligned(16)));  // 2 x 8 KB
  __shared__ u16 lB[2][TILE_N][TILE_K + 8] __attribute__((aligned(16)));  // 2 x 10 KB

  const int tid    = threadIdx.x;
  const int wave   = tid >> 5;
  const int lane   = tid & 31;
  const int laneM  = lane & 15;   // row (A/C) or col (B) within 16
  const int laneHi = lane >> 4;   // half-wave select per ISA layouts
  const int wr = wave >> 1;       // 0..3 : wave's 32-row strip
  const int wc = wave & 1;        // 0..1 : wave's 64-col strip
  const int m0 = blockIdx.y * TILE_M;
  const int n0 = blockIdx.x * TILE_N;

  v8f acc[2][4];
#pragma unroll
  for (int r = 0; r < 2; ++r)
#pragma unroll
    for (int i = 0; i < 4; ++i)
#pragma unroll
      for (int j = 0; j < 8; ++j) acc[r][i][j] = 0.0f;

  // staging assignments (exact fit: all dims divide evenly)
  const int arow  = tid >> 1;          // 0..127
  const int ahalf = (tid & 1) * 16;    // 0 or 16 (k offset)
  const int brow  = tid >> 3;          // 0..31  (k within tile)
  const int bseg  = (tid & 7) * 16;    // 0..112 (n offset)

  auto stageA = [&](int k0, int bufi) {
#if USE_TDM
    if (wave == 0)   // one TDM op per tile; EXEC-independent DMA into LDS
      tdm_load_tile_a(A + (size_t)m0 * Kd + k0,
                      (u32)(uintptr_t)&lA[bufi][0][0], Kd);
#else
    const uint4* ag = (const uint4*)(A + (size_t)(m0 + arow) * Kd + k0 + ahalf);
    uint4 a0v = ag[0], a1v = ag[1];
    *(uint4*)&lA[bufi][arow][ahalf]     = a0v;
    *(uint4*)&lA[bufi][arow][ahalf + 8] = a1v;
#endif
  };
  auto stageB = [&](int k0, int bufi) {
    // B tile transposed -> lB[n][k]: 16 bf16 per thread
    union { uint4 q; u16 e[8]; } b0, b1;
    const uint4* bg = (const uint4*)(Bw + (size_t)(k0 + brow) * N + n0 + bseg);
    b0.q = bg[0];
    b1.q = bg[1];
#pragma unroll
    for (int j = 0; j < 8; ++j) lB[bufi][bseg + j][brow]     = b0.e[j];
#pragma unroll
    for (int j = 0; j < 8; ++j) lB[bufi][bseg + 8 + j][brow] = b1.e[j];
  };
  auto tdm_fence = [&]() {
#if USE_TDM
    if (wave == 0) tdm_wait0();   // TDM write to LDS complete before barrier
#endif
  };

  stageA(0, 0);
  stageB(0, 0);
  tdm_fence();
  __syncthreads();

  int buf = 0;
  for (int k0 = 0; k0 < Kd; k0 += TILE_K) {
    if (k0 + TILE_K < Kd) {
      stageA(k0 + TILE_K, buf ^ 1);         // overlap with this step's WMMAs
      stageB(k0 + TILE_K, buf ^ 1);
      if (k0 + 2 * TILE_K < Kd) {           // prefetch tile k+2 -> global_prefetch_b8
        __builtin_prefetch(Bw + (size_t)(k0 + 2 * TILE_K + brow) * N + n0 + bseg, 0, 1);
#if !USE_TDM
        __builtin_prefetch(A + (size_t)(m0 + arow) * Kd + (k0 + 2 * TILE_K) + ahalf, 0, 1);
#endif
      }
    }

    // fragment loads (hoisted burst), then 8 back-to-back WMMAs
    union Frag { v16bf v; uint4 q[2]; };
    Frag fa[2], fb[4];
#pragma unroll
    for (int r = 0; r < 2; ++r) {
      // A frag (16x32): elems 0..15 -> K=hi*16+j (contiguous, matches B)
      const int ar = wr * 32 + r * 16 + laneM;
      fa[r].q[0] = *(const uint4*)&lA[buf][ar][laneHi * 16];
      fa[r].q[1] = *(const uint4*)&lA[buf][ar][laneHi * 16 + 8];
    }
#pragma unroll
    for (int i = 0; i < 4; ++i) {
      // B frag (32x16): col = laneM, K = laneHi*16 .. +15 (contiguous in lB row)
      const u16* bp = &lB[buf][wc * 64 + i * 16 + laneM][laneHi * 16];
      fb[i].q[0] = *(const uint4*)(bp);
      fb[i].q[1] = *(const uint4*)(bp + 8);
    }
#pragma unroll
    for (int r = 0; r < 2; ++r)
#pragma unroll
      for (int i = 0; i < 4; ++i)
        acc[r][i] = __builtin_amdgcn_wmma_f32_16x16x32_bf16(
            false, fa[r].v, false, fb[i].v, (short)0, acc[r][i], false, false);

    tdm_fence();
    __syncthreads();
    buf ^= 1;
  }

  // epilogue: C/D layout -> VGPR j, lane: row = hi*8 + j, col = laneM
#pragma unroll
  for (int r = 0; r < 2; ++r) {
    const int rowbase = m0 + wr * 32 + r * 16 + laneHi * 8;
#pragma unroll
    for (int i = 0; i < 4; ++i) {
      const int col = n0 + wc * 64 + i * 16 + laneM;
      const float bv = bias ? bias[col] : 0.0f;
#pragma unroll
      for (int j = 0; j < 8; ++j) {
        const int row = rowbase + j;
        float v = acc[r][i][j] + bv;
        if (relu) v = fmaxf(v, 0.0f);
        const size_t o = (size_t)row * N + col;
        if (outb) outb[o] = f2bf(v);
        if (outf) outf[o] = v;
      }
    }
  }
}

// ---------------------------------------------------------------------------
// VQ-EMA kernels
// ---------------------------------------------------------------------------
__global__ void zero_kernel(float* __restrict__ p, size_t n) {
  size_t i  = (size_t)blockIdx.x * blockDim.x + threadIdx.x;
  size_t st = (size_t)gridDim.x * blockDim.x;
  for (; i < n; i += st) p[i] = 0.0f;
}

// ||e_k||^2 for each codebook column (embeddings are [D, K])
__global__ void enorm_kernel(const float* __restrict__ emb, float* __restrict__ en) {
  int k = blockIdx.x * blockDim.x + threadIdx.x;
  if (k >= KC) return;
  float s = 0.0f;
  for (int d = 0; d < DL; ++d) {
    float e = emb[(size_t)d * KC + k];
    s += e * e;
  }
  en[k] = s;
}

// argmin_k ( ||e_k||^2 - 2 * (z @ E)[b,k] )  -- ||z||^2 is constant per row
__global__ __launch_bounds__(256) void argmin_kernel(
    const float* __restrict__ scores, const float* __restrict__ en,
    int* __restrict__ idx) {
  const int wave = threadIdx.x >> 5, lane = threadIdx.x & 31;
  const int b = blockIdx.x * 8 + wave;
  const float* s = scores + (size_t)b * KC;
  float best = 3.4e38f;
  int bi = 0x7FFFFFFF;
  for (int k = lane; k < KC; k += 32) {
    float d = en[k] - 2.0f * s[k];
    if (d < best || (d == best && k < bi)) { best = d; bi = k; }
  }
  for (int off = 16; off > 0; off >>= 1) {   // wave32 reduction
    float ob = __shfl_xor(best, off, 32);
    int   oi = __shfl_xor(bi,   off, 32);
    if (ob < best || (ob == best && oi < bi)) { best = ob; bi = oi; }
  }
  if (lane == 0) idx[b] = bi;
}

__global__ void counts_kernel(const int* __restrict__ idx, float* __restrict__ counts) {
  int b = blockIdx.x * blockDim.x + threadIdx.x;
  if (b < BATCH) atomicAdd(&counts[idx[b]], 1.0f);
}

// q_st = e[:, idx]; accumulate sum((q - z)^2) and dw_acc[d,k] += z[b,d]
__global__ __launch_bounds__(256) void qst_kernel(
    const float* __restrict__ z, const float* __restrict__ emb,
    const int* __restrict__ idx, float* __restrict__ qst,
    float* __restrict__ lossacc, float* __restrict__ dwacc) {
  __shared__ float red[256];
  const size_t n  = (size_t)BATCH * DL;
  const size_t st = (size_t)gridDim.x * blockDim.x;
  float ls = 0.0f;
  for (size_t i = (size_t)blockIdx.x * blockDim.x + threadIdx.x; i < n; i += st) {
    int b = (int)(i / DL), d = (int)(i % DL);
    int k = idx[b];
    float q  = emb[(size_t)d * KC + k];
    float zz = z[i];
    qst[i] = q;                         // z + sg(q - z) == q numerically
    float df = q - zz;
    ls += df * df;
    atomicAdd(&dwacc[(size_t)d * KC + k], zz);   // (z^T @ enc) scatter
  }
  red[threadIdx.x] = ls;
  __syncthreads();
  for (int s2 = 128; s2 > 0; s2 >>= 1) {
    if ((int)threadIdx.x < s2) red[threadIdx.x] += red[threadIdx.x + s2];
    __syncthreads();
  }
  if (threadIdx.x == 0) atomicAdd(lossacc, red[0]);
}

// cluster-size EMA, Laplace smoothing, perplexity, loss scalar
__global__ __launch_bounds__(512) void stats_kernel(
    const float* __restrict__ counts, const float* __restrict__ ema_cs,
    const float* __restrict__ lossacc,
    float* __restrict__ out_loss, float* __restrict__ out_perp,
    float* __restrict__ out_cs) {
  __shared__ float sn[KC], sh[KC];
  const int k = threadIdx.x;
  float c  = counts[k];
  float cs = DECAY * ema_cs[k] + (1.0f - DECAY) * c;
  float p  = c / (float)BATCH;
  sn[k] = cs;
  sh[k] = -p * logf(p + 1e-10f);
  __syncthreads();
  for (int s2 = KC / 2; s2 > 0; s2 >>= 1) {
    if (k < s2) { sn[k] += sn[k + s2]; sh[k] += sh[k + s2]; }
    __syncthreads();
  }
  const float nsum = sn[0];
  out_cs[k] = (cs + EPSF) / (nsum + (float)KC * EPSF) * nsum;
  if (k == 0) {
    out_loss[0] = CCOST * lossacc[0] / (float)((size_t)BATCH * DL);
    out_perp[0] = expf(sh[0]);
  }
}

__global__ void newemb_kernel(const float* __restrict__ ema_dw,
                              const float* __restrict__ dwacc,
                              const float* __restrict__ cs_stable,
                              float* __restrict__ out_emb) {
  size_t i = (size_t)blockIdx.x * blockDim.x + threadIdx.x;
  if (i >= (size_t)DL * KC) return;
  int k = (int)(i % KC);
  out_emb[i] = (DECAY * ema_dw[i] + (1.0f - DECAY) * dwacc[i]) / cs_stable[k];
}

// ---------------------------------------------------------------------------
extern "C" void kernel_launch(void* const* d_in, const int* in_sizes, int n_in,
                              void* d_out, int out_size, void* d_ws, size_t ws_size,
                              hipStream_t stream) {
  (void)in_sizes; (void)n_in; (void)out_size; (void)ws_size;

  const float* obs  = (const float*)d_in[0];
  const float* actn = (const float*)d_in[1];
  const float* nobs = (const float*)d_in[2];
  const float* rew  = (const float*)d_in[3];
  const float* term = (const float*)d_in[4];
  const float* w[4]  = {(const float*)d_in[5], (const float*)d_in[7],
                        (const float*)d_in[9], (const float*)d_in[11]};
  const float* bb[4] = {(const float*)d_in[6], (const float*)d_in[8],
                        (const float*)d_in[10], (const float*)d_in[12]};
  const float* wl     = (const float*)d_in[13];
  const float* bl     = (const float*)d_in[14];
  const float* emb    = (const float*)d_in[15];
  const float* ema_cs = (const float*)d_in[16];
  const float* ema_dw = (const float*)d_in[17];

  // workspace layout
  char* ws = (char*)d_ws;
  size_t off = 0;
  auto walloc = [&](size_t bytes) -> void* {
    void* p = ws + off;
    off = (off + bytes + 255) & ~(size_t)255;
    return p;
  };
  u16*   X    = (u16*)walloc((size_t)BATCH * DINP * 2);   // packed input, bf16
  u16*   a0b  = (u16*)walloc((size_t)BATCH * HID * 2);    // activation ping
  u16*   a1b  = (u16*)walloc((size_t)BATCH * HID * 2);    // activation pong
  u16*   Wc   = (u16*)walloc((size_t)HID * HID * 2);      // converted weight (reused)
  float* zf   = (float*)walloc((size_t)BATCH * DL * 4);   // z fp32
  float* sc   = (float*)walloc((size_t)BATCH * KC * 4);   // z @ E scores
  int*   idx  = (int*)walloc((size_t)BATCH * 4);
  float* en   = (float*)walloc((size_t)KC * 4);
  float* rz   = (float*)walloc((KC + 64 + (size_t)DL * KC) * 4); // zeroed region
  float* counts  = rz;
  float* lossacc = rz + KC;
  float* dwacc   = rz + KC + 64;

  // output layout: q_st [B,DL] | loss | perplexity | new_embeddings [DL,KC] | cs_stable [KC]
  float* out    = (float*)d_out;
  float* o_q    = out;
  float* o_loss = out + (size_t)BATCH * DL;
  float* o_perp = o_loss + 1;
  float* o_emb  = o_perp + 1;
  float* o_cs   = o_emb + (size_t)DL * KC;

  auto cvt = [&](const float* Wp, int Krows, int Kpad, int N) {
    size_t n = (size_t)Kpad * N;
    cvt_w_kernel<<<(unsigned)((n + 255) / 256), 256, 0, stream>>>(Wp, Wc, Krows, Kpad, N);
  };
  auto gemm = [&](const u16* Ab, const float* bias, u16* ob, float* of,
                  int M, int Kd, int N, int relu) {
    dim3 g(N / TILE_N, M / TILE_M);
    gemm_bf16_wmma<<<g, 256, 0, stream>>>(Ab, Wc, bias, ob, of, M, Kd, N, relu);
  };

  { size_t n = (size_t)BATCH * DINP;
    pack_x_kernel<<<(unsigned)((n + 255) / 256), 256, 0, stream>>>(
        obs, actn, nobs, rew, term, X); }

  // MLP encoder (bf16 WMMA, f32 accumulate)
  cvt(w[0], DIN, DINP, HID); gemm(X,   bb[0], a0b, nullptr, BATCH, DINP, HID, 1);
  cvt(w[1], HID, HID, HID);  gemm(a0b, bb[1], a1b, nullptr, BATCH, HID,  HID, 1);
  cvt(w[2], HID, HID, HID);  gemm(a1b, bb[2], a0b, nullptr, BATCH, HID,  HID, 1);
  cvt(w[3], HID, HID, HID);  gemm(a0b, bb[3], a1b, nullptr, BATCH, HID,  HID, 1);
  // final projection: dual-write z (bf16 for VQ score GEMM, fp32 for loss/EMA)
  cvt(wl, HID, HID, DL);     gemm(a1b, bl, a0b, zf, BATCH, HID, DL, 0);
  // VQ distance GEMM: scores = z @ E   (E converted [DL,KC] -> bf16)
  cvt(emb, DL, DL, KC);      gemm(a0b, nullptr, nullptr, sc, BATCH, DL, KC, 0);

  // VQ-EMA tail
  zero_kernel<<<1024, 256, 0, stream>>>(rz, KC + 64 + (size_t)DL * KC);
  enorm_kernel<<<(KC + 255) / 256, 256, 0, stream>>>(emb, en);
  argmin_kernel<<<BATCH / 8, 256, 0, stream>>>(sc, en, idx);
  counts_kernel<<<BATCH / 256, 256, 0, stream>>>(idx, counts);
  qst_kernel<<<2048, 256, 0, stream>>>(zf, emb, idx, o_q, lossacc, dwacc);
  stats_kernel<<<1, KC, 0, stream>>>(counts, ema_cs, lossacc, o_loss, o_perp, o_cs);
  { size_t n = (size_t)DL * KC;
    newemb_kernel<<<(unsigned)((n + 255) / 256), 256, 0, stream>>>(
        ema_dw, dwacc, o_cs, o_emb); }
}
